// CriticRNN_4664334484024
// MI455X (gfx1250) — matrix operand order repaired and checked by reading, not compiled, not run on hardware
//
#include <hip/hip_runtime.h>

typedef __bf16 v16bf __attribute__((ext_vector_type(16)));
typedef float  v8f   __attribute__((ext_vector_type(8)));

union frag16 { v16bf v; uint4 u[2]; };   // 16 bf16 = 32B = 2 x b128

// Model constants: T=128, NE=64, NA=8, B=512, OBS=64, D=128, CH=128, VH=256

// ---------------------------------------------------------------------------
// CDNA5 async global->LDS copy (ASYNCcnt-tracked, bypasses VGPRs).
// Generic shared pointers keep the LDS byte offset in addr[31:0].
// ---------------------------------------------------------------------------
__device__ __forceinline__ unsigned lds_u32(const void* p) {
  return (unsigned)(unsigned long long)p;
}
__device__ __forceinline__ void async_load_b128(unsigned lds_off, const void* gptr) {
  asm volatile("global_load_async_to_lds_b128 %0, %1, off"
               :: "v"(lds_off), "v"(gptr)
               : "memory");
}
__device__ __forceinline__ void wait_async() {
#if __has_builtin(__builtin_amdgcn_s_wait_asynccnt)
  __builtin_amdgcn_s_wait_asynccnt(0);
#else
  asm volatile("s_wait_asynccnt 0x0" ::: "memory");
#endif
}

// ---------------------------------------------------------------------------
// Weight conversion, f32 -> bf16 TRANSPOSED: src (K x N) -> dst (N x K).
// Makes every WMMA B-fragment a pair of contiguous b128 loads.
// ---------------------------------------------------------------------------
__global__ __launch_bounds__(256) void k_cvt_t(const float* __restrict__ src,
                                               __bf16* __restrict__ dst,
                                               int K, int N) {
  int i = blockIdx.x * 256 + threadIdx.x;
  if (i < K * N) {
    int k = i / N, n = i - k * N;
    dst[(size_t)n * K + k] = (__bf16)src[i];
  }
}

// ---------------------------------------------------------------------------
// Tiled WMMA GEMM: out = epilogue(A @ B + bias), B given TRANSPOSED (N x K).
//   K, K1 compile-time: K-loop fully unrolls into K/32 v_wmma per tile.
//   A: (M x K) bf16 row-major, optionally split [A1 | A2] at K1, or f32 (A1f).
//   mode: 0 none, 1 relu, 2 e_new = (e_old + relu(acc+bias)) * alive
// Block = 64 rows x 128 cols, 8 waves: wave w owns cols n0+16w for all 4
// 16-row tiles, so each B fragment (read from L2) is reused 4x. A slab is
// async-copied to LDS and shared by all waves.
// ---------------------------------------------------------------------------
template <int K, int K1>
__global__ __launch_bounds__(256) void k_gemm(
    const __bf16* __restrict__ A1, const __bf16* __restrict__ A2,
    const float* __restrict__ A1f,
    const __bf16* __restrict__ BT, int ldbt,
    const float* __restrict__ bias,
    float* __restrict__ outf, __bf16* __restrict__ outbf, int ldc,
    int mode, const float* __restrict__ e_old,
    const unsigned char* __restrict__ dones) {
  __shared__ __align__(16) __bf16 Asl[64][K + 8];

  const int tid  = threadIdx.x;
  const int wave = tid >> 5, lane = tid & 31;
  const int m0 = blockIdx.x << 6;
  const int n0 = blockIdx.y << 7;

  // Stage the 64-row A slab
  if (A1f) {  // f32 source (K1 == K): convert in flight
    for (int i = tid; i < 64 * (K / 4); i += 256) {
      int r = i / (K / 4), q = (i - r * (K / 4)) * 4;
      float4 v = *(const float4*)(A1f + (size_t)(m0 + r) * K + q);
      __bf16* p = &Asl[r][q];
      p[0] = (__bf16)v.x; p[1] = (__bf16)v.y; p[2] = (__bf16)v.z; p[3] = (__bf16)v.w;
    }
  } else {    // bf16 source: async DMA straight into LDS
    for (int i = tid; i < 64 * (K / 8); i += 256) {
      int r = i / (K / 8), q = (i - r * (K / 8)) * 8;
      const __bf16* src = (q < K1)
          ? A1 + (size_t)(m0 + r) * K1 + q
          : A2 + (size_t)(m0 + r) * (K - K1) + (q - K1);
      async_load_b128(lds_u32(&Asl[r][q]), src);
    }
    wait_async();
  }
  __syncthreads();

  const int arow = lane & 15;
  const int akb  = (lane < 16) ? 0 : 8;    // A: K 0-7/16-23 vs 8-15/24-31
  const int bkh  = (lane < 16) ? 0 : 8;    // B: K 0-7/16-23 vs 8-15/24-31
  const int ncol = (wave << 4) + (lane & 15);
  const __bf16* bcol = BT + (size_t)(n0 + ncol) * ldbt;

  v8f acc[4] = {{}, {}, {}, {}};
#pragma unroll
  for (int k0 = 0; k0 < K; k0 += 32) {
    frag16 bf_;
    const uint4* bp = (const uint4*)(bcol + k0 + bkh);
    bf_.u[0] = bp[0];
    bf_.u[1] = bp[2];
#pragma unroll
    for (int mt = 0; mt < 4; ++mt) {
      frag16 af;
      af.u[0] = *(const uint4*)&Asl[(mt << 4) + arow][k0 + akb];
      af.u[1] = *(const uint4*)&Asl[(mt << 4) + arow][k0 + 16 + akb];
      acc[mt] = __builtin_amdgcn_wmma_f32_16x16x32_bf16(false, af.v, false, bf_.v,
                                                        (short)0, acc[mt], false, false);
    }
  }

  const int coln = n0 + ncol;
#pragma unroll
  for (int mt = 0; mt < 4; ++mt) {
#pragma unroll
    for (int r = 0; r < 8; ++r) {
      int rl = (mt << 4) + r + ((lane < 16) ? 0 : 8);
      size_t m = (size_t)(m0 + rl);
      float v = acc[mt][r];
      if (bias) v += bias[coln];
      if (mode >= 1) v = fmaxf(v, 0.0f);
      if (mode == 2) {
        float al = dones[m] ? 0.0f : 1.0f;
        v = (e_old[m * ldc + coln] + v) * al;
      }
      if (outf)  outf[m * ldc + coln] = v;
      if (outbf) outbf[m * ldc + coln] = (__bf16)v;
    }
  }
}

// ---------------------------------------------------------------------------
// Persistent GRU scan. 32 workgroups own 16-row slabs of h; loop T=128 steps.
// WiT+WhT (192KB bf16) are async-copied ONCE into dynamic LDS (CDNA5 allows
// up to 320KB/WG) so the serial path never re-reads weights from L2.
// Wave w computes the three gi and three gh tiles for cols [16w,16w+16) so
// the gate math is register-local. h kept in LDS as f32 + bf16 shadow.
// ---------------------------------------------------------------------------
__global__ __launch_bounds__(256) void k_gru(
    const float* __restrict__ h_in, const __bf16* __restrict__ emb,
    const __bf16* __restrict__ WiT, const __bf16* __restrict__ WhT,
    const float* __restrict__ bi, const float* __restrict__ bhn,
    const unsigned char* __restrict__ dones,
    float* __restrict__ e_f, __bf16* __restrict__ e_b,
    float* __restrict__ h_out) {
  extern __shared__ __bf16 wlds[];                 // [384*128] WiT | [384*128] WhT
  __bf16* Wl  = wlds;
  __bf16* Whl = wlds + 384 * 128;
  __shared__ float  hsl[16][132];
  __shared__ __align__(16) __bf16 hbsl[16][136];
  __shared__ __align__(16) __bf16 xsl[16][136];
  const int tid = threadIdx.x, wave = tid >> 5, lane = tid & 31;
  const int b0 = blockIdx.x << 4;
  const int arow = lane & 15, akb = (lane < 16) ? 0 : 8, bkh = (lane < 16) ? 0 : 8;
  const int d = (wave << 4) + (lane & 15);

  // One-time async preload of both weight matrices into LDS
  for (int i = tid; i < (384 * 128) / 8; i += 256) {
    async_load_b128(lds_u32(&Wl[i * 8]),  WiT + (size_t)i * 8);
    async_load_b128(lds_u32(&Whl[i * 8]), WhT + (size_t)i * 8);
  }
  for (int i = tid; i < 16 * 128; i += 256) {
    float v = h_in[(size_t)(b0 + (i >> 7)) * 128 + (i & 127)];
    hsl[i >> 7][i & 127]  = v;
    hbsl[i >> 7][i & 127] = (__bf16)v;
  }
  wait_async();
  __syncthreads();

  for (int t = 0; t < 128; ++t) {
    // stage x_t slab via async DMA; reset h where done
    {
      int r = tid >> 4, q = (tid & 15) * 8;   // 256 threads = 16 rows x 16 vecs
      async_load_b128(lds_u32(&xsl[r][q]),
                      emb + ((size_t)t * 512 + b0 + r) * 128 + q);
      if (t + 1 < 128)   // pull next step's tile toward L2/L0 early
        __builtin_prefetch(emb + ((size_t)(t + 1) * 512 + b0 + r) * 128 + q, 0, 1);
    }
    for (int i = tid; i < 16 * 128; i += 256) {
      int r = i >> 7, k = i & 127;
      if (dones[t * 512 + b0 + r]) { hsl[r][k] = 0.0f; hbsl[r][k] = (__bf16)0.0f; }
    }
    wait_async();
    __syncthreads();

    v8f gi[3] = {{}, {}, {}}, gh[3] = {{}, {}, {}};
#pragma unroll
    for (int k0 = 0; k0 < 128; k0 += 32) {
      frag16 ax, ah;
      ax.u[0] = *(const uint4*)&xsl[arow][k0 + akb];
      ax.u[1] = *(const uint4*)&xsl[arow][k0 + 16 + akb];
      ah.u[0] = *(const uint4*)&hbsl[arow][k0 + akb];
      ah.u[1] = *(const uint4*)&hbsl[arow][k0 + 16 + akb];
#pragma unroll
      for (int p = 0; p < 3; ++p) {
        int c = (p << 7) + d;   // gate column in the 384-wide weights
        frag16 bwi, bwh;
        const uint4* wip = (const uint4*)(Wl  + (size_t)c * 128 + k0 + bkh);
        const uint4* whp = (const uint4*)(Whl + (size_t)c * 128 + k0 + bkh);
        bwi.u[0] = wip[0]; bwi.u[1] = wip[2];
        bwh.u[0] = whp[0]; bwh.u[1] = whp[2];
        gi[p] = __builtin_amdgcn_wmma_f32_16x16x32_bf16(false, ax.v, false, bwi.v,
                                                        (short)0, gi[p], false, false);
        gh[p] = __builtin_amdgcn_wmma_f32_16x16x32_bf16(false, ah.v, false, bwh.v,
                                                        (short)0, gh[p], false, false);
      }
    }

    float hnew[8];
#pragma unroll
    for (int r = 0; r < 8; ++r) {
      int rl = r + ((lane < 16) ? 0 : 8);
      float ir  = gi[0][r] + bi[d];
      float iz  = gi[1][r] + bi[128 + d];
      float in_ = gi[2][r] + bi[256 + d];
      float rg = 1.0f / (1.0f + __expf(-(ir + gh[0][r])));
      float zg = 1.0f / (1.0f + __expf(-(iz + gh[1][r])));
      float ng = tanhf(in_ + rg * (gh[2][r] + bhn[d]));
      hnew[r] = (1.0f - zg) * ng + zg * hsl[rl][d];
    }
    __syncthreads();   // all waves done reading h before overwrite
#pragma unroll
    for (int r = 0; r < 8; ++r) {
      int rl = r + ((lane < 16) ? 0 : 8);
      hsl[rl][d]  = hnew[r];
      hbsl[rl][d] = (__bf16)hnew[r];
      size_t m = (size_t)t * 512 + b0 + rl;
      float al = dones[m] ? 0.0f : 1.0f;
      float ev = hnew[r] * al;        // e = rnn_out * alive
      e_f[m * 128 + d] = ev;
      e_b[m * 128 + d] = (__bf16)ev;
    }
    __syncthreads();
  }

  for (int i = tid; i < 16 * 128; i += 256)
    h_out[(size_t)(b0 + (i >> 7)) * 128 + (i & 127)] = hsl[i >> 7][i & 127];
}

// ---------------------------------------------------------------------------
// Pairwise coupling gate: C[g,i,j] = sigmoid(sum_h relu(ai+aj+b)*w + b0),
// masked by identity and alive(j). One block per (t,env); 64 threads = pairs.
// ---------------------------------------------------------------------------
__global__ __launch_bounds__(64) void k_couple(
    const float* __restrict__ ai, const float* __restrict__ aj,
    const float* __restrict__ cb, const float* __restrict__ cw,
    const float* __restrict__ cob, const unsigned char* __restrict__ dones,
    float* __restrict__ C) {
  int g = blockIdx.x;                      // g = t*64 + env; row m = g*8 + agent
  __shared__ float ais[8][128], ajs[8][128], ws[128], bs[128];
  int tid = threadIdx.x;
  for (int i = tid; i < 8 * 128; i += 64) {
    int r = i >> 7, h = i & 127;
    size_t m = (size_t)g * 8 + r;
    ais[r][h] = ai[m * 128 + h];
    ajs[r][h] = aj[m * 128 + h];
  }
  for (int i = tid; i < 128; i += 64) { ws[i] = cw[i]; bs[i] = cb[i]; }
  __syncthreads();
  int i = tid >> 3, j = tid & 7;
  float s = cob[0];
  for (int h = 0; h < 128; ++h)
    s += fmaxf(ais[i][h] + ajs[j][h] + bs[h], 0.0f) * ws[h];
  float c = 1.0f / (1.0f + __expf(-s));
  if (i == j) c = 0.0f;
  if (dones[(size_t)g * 8 + j]) c = 0.0f;
  C[(size_t)g * 64 + i * 8 + j] = c;
}

// ---------------------------------------------------------------------------
// context[i,:] = sum_j C[i,j] * e[j,:]  (8x8 @ 8x128 per (t,env)) -> bf16
// ---------------------------------------------------------------------------
__global__ __launch_bounds__(128) void k_context(
    const float* __restrict__ C, const float* __restrict__ e,
    __bf16* __restrict__ ctx) {
  int g = blockIdx.x;
  __shared__ float es[8][128];
  __shared__ float Cs[64];
  int tid = threadIdx.x;
  for (int i = tid; i < 1024; i += 128) {
    int r = i >> 7, h = i & 127;
    es[r][h] = e[((size_t)g * 8 + r) * 128 + h];
  }
  if (tid < 64) Cs[tid] = C[(size_t)g * 64 + tid];
  __syncthreads();
  int dcol = tid;
#pragma unroll
  for (int i = 0; i < 8; ++i) {
    float s = 0.0f;
#pragma unroll
    for (int j = 0; j < 8; ++j) s += Cs[i * 8 + j] * es[j][dcol];
    ctx[((size_t)g * 8 + i) * 128 + dcol] = (__bf16)s;
  }
}

// ---------------------------------------------------------------------------
// values[m] = dot(v2[m,:256], vo) + b   (one wave per row)
// ---------------------------------------------------------------------------
__global__ __launch_bounds__(256) void k_vout(const __bf16* __restrict__ v2,
                                              const float* __restrict__ vw,
                                              const float* __restrict__ vb,
                                              float* __restrict__ out) {
  int wave = threadIdx.x >> 5, lane = threadIdx.x & 31;
  size_t m = (size_t)blockIdx.x * 8 + wave;
  float s = 0.0f;
  for (int h = lane; h < 256; h += 32) s += (float)v2[m * 256 + h] * vw[h];
#pragma unroll
  for (int off = 16; off; off >>= 1) s += __shfl_xor(s, off, 32);
  if (lane == 0) out[m] = s + vb[0];
}

// ---------------------------------------------------------------------------
extern "C" void kernel_launch(void* const* d_in, const int* in_sizes, int n_in,
                              void* d_out, int out_size, void* d_ws, size_t ws_size,
                              hipStream_t stream) {
  (void)in_sizes; (void)n_in; (void)out_size; (void)ws_size;
  const float* hidden = (const float*)d_in[0];
  const float* obs    = (const float*)d_in[1];
  const unsigned char* dones = (const unsigned char*)d_in[2];
  const float* e1w = (const float*)d_in[3];
  const float* e1b = (const float*)d_in[4];
  const float* e2w = (const float*)d_in[5];
  const float* e2b = (const float*)d_in[6];
  const float* gWi = (const float*)d_in[7];
  const float* gbi = (const float*)d_in[8];
  const float* gWh = (const float*)d_in[9];
  const float* gbhn = (const float*)d_in[10];
  const float* chw = (const float*)d_in[11];
  const float* chb = (const float*)d_in[12];
  const float* cow = (const float*)d_in[13];
  const float* cob = (const float*)d_in[14];
  const float* uhw = (const float*)d_in[15];
  const float* uhb = (const float*)d_in[16];
  const float* uow = (const float*)d_in[17];
  const float* uob = (const float*)d_in[18];
  const float* v1w = (const float*)d_in[19];
  const float* v1b = (const float*)d_in[20];
  const float* v2w = (const float*)d_in[21];
  const float* v2b = (const float*)d_in[22];
  const float* vow = (const float*)d_in[23];
  const float* vob = (const float*)d_in[24];

  float* out_hidden = (float*)d_out;          // (512,128)
  float* out_values = (float*)d_out + 65536;  // (128,512)

  const size_t M = 65536;  // T*B rows
  char* wp = (char*)d_ws;
  auto alloc = [&](size_t bytes) {
    void* p = (void*)wp;
    wp += (bytes + 255) & ~(size_t)255;
    return p;
  };
  // transposed bf16 weights (all L2-resident)
  __bf16* e1wT = (__bf16*)alloc(128 * 64 * 2);
  __bf16* e2wT = (__bf16*)alloc(128 * 128 * 2);
  __bf16* gWiT = (__bf16*)alloc(384 * 128 * 2);
  __bf16* gWhT = (__bf16*)alloc(384 * 128 * 2);
  __bf16* chwT = (__bf16*)alloc(128 * 256 * 2);  // rows n: [W1T(n,:) | W2T(n,:)]
  __bf16* uhwT = (__bf16*)alloc(128 * 256 * 2);
  __bf16* uowT = (__bf16*)alloc(128 * 128 * 2);
  __bf16* v1wT = (__bf16*)alloc(256 * 128 * 2);
  __bf16* v2wT = (__bf16*)alloc(256 * 256 * 2);
  // activations
  __bf16* h1_b  = (__bf16*)alloc(M * 128 * 2);
  __bf16* emb_b = (__bf16*)alloc(M * 128 * 2);
  float*  e_f   = (float*) alloc(M * 128 * 4);
  __bf16* e_b   = (__bf16*)alloc(M * 128 * 2);
  float*  ai_f  = (float*) alloc(M * 128 * 4);
  float*  aj_f  = (float*) alloc(M * 128 * 4);
  float*  C_f   = (float*) alloc((size_t)8192 * 64 * 4);
  __bf16* ctx_b = (__bf16*)alloc(M * 128 * 2);
  __bf16* g1_b  = (__bf16*)alloc(M * 128 * 2);
  __bf16* v1_b  = (__bf16*)alloc(M * 256 * 2);
  __bf16* v2_b  = (__bf16*)alloc(M * 256 * 2);

  auto cvtT = [&](const float* s, __bf16* dst, int K, int N) {
    k_cvt_t<<<(K * N + 255) / 256, 256, 0, stream>>>(s, dst, K, N);
  };
  cvtT(e1w, e1wT, 64, 128);
  cvtT(e2w, e2wT, 128, 128);
  cvtT(gWi, gWiT, 128, 384);
  cvtT(gWh, gWhT, 128, 384);
  cvtT(chw, chwT, 256, 128);
  cvtT(uhw, uhwT, 256, 128);
  cvtT(uow, uowT, 128, 128);
  cvtT(v1w, v1wT, 128, 256);
  cvtT(v2w, v2wT, 256, 256);

  // Embedding MLP
  k_gemm<64, 64><<<dim3(M / 64, 1), 256, 0, stream>>>(nullptr, nullptr, obs,
      e1wT, 64, e1b, nullptr, h1_b, 128, 1, nullptr, nullptr);
  k_gemm<128, 128><<<dim3(M / 64, 1), 256, 0, stream>>>(h1_b, nullptr, nullptr,
      e2wT, 128, e2b, nullptr, emb_b, 128, 1, nullptr, nullptr);

  // Sequential GRU scan: 32 persistent blocks, weights resident in 192KB LDS
  k_gru<<<32, 256, 384 * 128 * 2 * 2, stream>>>(hidden, emb_b, gWiT, gWhT,
      gbi, gbhn, dones, e_f, e_b, out_hidden);

  // Coupling iterations
  for (int it = 0; it < 2; ++it) {
    k_gemm<128, 128><<<dim3(M / 64, 1), 256, 0, stream>>>(e_b, nullptr, nullptr,
        chwT, 256, nullptr, ai_f, nullptr, 128, 0, nullptr, nullptr);
    k_gemm<128, 128><<<dim3(M / 64, 1), 256, 0, stream>>>(e_b, nullptr, nullptr,
        chwT + 128, 256, nullptr, aj_f, nullptr, 128, 0, nullptr, nullptr);
    k_couple<<<8192, 64, 0, stream>>>(ai_f, aj_f, chb, cow, cob, dones, C_f);
    k_context<<<8192, 128, 0, stream>>>(C_f, e_f, ctx_b);
    // delta1 = relu([e | ctx] @ [U1;U2] + b)   (K=256 via split A)
    k_gemm<256, 128><<<dim3(M / 64, 1), 256, 0, stream>>>(e_b, ctx_b, nullptr,
        uhwT, 256, uhb, nullptr, g1_b, 128, 1, nullptr, nullptr);
    // e = (e + relu(delta1 @ Uo + b)) * alive   (fused residual epilogue)
    k_gemm<128, 128><<<dim3(M / 64, 1), 256, 0, stream>>>(g1_b, nullptr, nullptr,
        uowT, 128, uob, e_f, e_b, 128, 2, e_f, dones);
  }

  // Value head
  k_gemm<128, 128><<<dim3(M / 64, 2), 256, 0, stream>>>(e_b, nullptr, nullptr,
      v1wT, 128, v1b, nullptr, v1_b, 256, 1, nullptr, nullptr);
  k_gemm<256, 256><<<dim3(M / 64, 2), 256, 0, stream>>>(v1_b, nullptr, nullptr,
      v2wT, 256, v2b, nullptr, v2_b, 256, 1, nullptr, nullptr);
  k_vout<<<8192, 256, 0, stream>>>(v2_b, vow, vob, out_values);
}
